// TransformerBlock_43585328120535
// MI455X (gfx1250) — compile-verified, matrix operand order checked by reading, not run on hardware
//
#include <hip/hip_runtime.h>
#include <hip/hip_bf16.h>

// ---------------------------------------------------------------------------
// TransformerBlock on MI455X (gfx1250): f16-operand / f32-accumulate WMMA,
// async global->LDS (ASYNCcnt) double-buffered GEMM pipeline.
// ---------------------------------------------------------------------------

typedef _Float16 h16;
typedef __attribute__((ext_vector_type(16))) _Float16 v16h;
typedef __attribute__((ext_vector_type(8)))  float    v8f;

#define B_SZ   2
#define S_LEN  2048
#define E_DIM  2048
#define H_NUM  16
#define D_HEAD 128
#define M_TOT  (B_SZ * S_LEN)   // 4096

// ------------------------- WMMA fragment helpers ---------------------------
// 16-bit A-matrix 16x32 layout (ISA 7.12.2): lane%16 = row M; half = lane/16.
// VGPR j (as half-pair 2j,2j+1) holds K = (j<4?0:16) + half*8 + 2*(j&3) .. +1
__device__ __forceinline__ int frag_kbase(int j, int hlf) {
    return ((j & 4) << 2) + hlf * 8 + ((j & 3) << 1);
}

// p points at the start of the (M or N) row; K is contiguous with unit stride.
__device__ __forceinline__ v16h load_frag_rowmajor(const h16* p, int hlf) {
    v16h f;
#pragma unroll
    for (int j = 0; j < 8; ++j) {
        int kb = frag_kbase(j, hlf);
        f[2 * j]     = p[kb];
        f[2 * j + 1] = p[kb + 1];
    }
    return f;
}

// ----------------- CDNA5 async global->LDS copy (ASYNCcnt) ----------------
// Portable across ROCm7.2 / clang-23 toolchains via inline asm (the clang
// builtin arity differs between them). vdst = LDS byte address (low 32 bits
// of the generic shared pointer), vaddr = 64-bit global address.
__device__ __forceinline__ void async_copy_b128(unsigned lds_off,
                                                unsigned long long gaddr) {
    asm volatile("global_load_async_to_lds_b128 %0, %1, off"
                 :: "v"(lds_off), "v"(gaddr)
                 : "memory");
}
__device__ __forceinline__ void wait_async0() {
    asm volatile("s_wait_asynccnt 0x0" ::: "memory");
}

__device__ __forceinline__ float block_reduce_sum256(float v, float* red) {
    int t = threadIdx.x;
    red[t] = v;
    __syncthreads();
#pragma unroll
    for (int st = 128; st > 0; st >>= 1) {
        if (t < st) red[t] += red[t + st];
        __syncthreads();
    }
    float r = red[0];
    __syncthreads();
    return r;
}

// --------------------------- f32 -> f16 convert ----------------------------
__global__ __launch_bounds__(256) void cvt_f32_to_f16_kernel(
    const float* __restrict__ src, h16* __restrict__ dst, int n) {
    int i = blockIdx.x * 256 + threadIdx.x;
    int stride = gridDim.x * 256;
    for (; i < n; i += stride) dst[i] = (h16)src[i];
}

// ------------------------------ WMMA GEMM ----------------------------------
// C(MxN) = alpha * A(MxK, f16 row-major) * Bw(NxK, f16 row-major)^T
// out_f16 != 0 -> write Ch (f16), else write Cf (f32).
// Double-buffered LDS; tiles filled by GLOBAL_LOAD_ASYNC_TO_LDS_B128.
#define BM 128
#define BN 128
#define BK 32
#define LDT (BK + 8)   // padded LDS row (halves) to avoid bank conflicts

__global__ __launch_bounds__(256, 2) void gemm_wmma_kernel(
    const h16* __restrict__ A, const h16* __restrict__ Bw,
    float* __restrict__ Cf, h16* __restrict__ Ch,
    int M, int N, int K, float alpha, int out_f16) {
    __shared__ h16 As[2][BM * LDT];
    __shared__ h16 Bs[2][BM * LDT];

    const int tid  = threadIdx.x;
    const int lane = tid & 31;
    const int wave = tid >> 5;
    const int hlf  = lane >> 4;
    const int l16  = lane & 15;
    const int m0   = blockIdx.y * BM;
    const int n0   = blockIdx.x * BN;
    const int wm   = (wave & 3) * 32;   // 4 waves along M (32 rows each)
    const int wn   = (wave >> 2) * 64;  // 2 waves along N (64 cols each)

    v8f acc[2][4] = {};

    // cooperative fill mapping: 2 threads per tile row, 16 halves (32B) each
    const int lr = tid >> 1;
    const int lc = (tid & 1) * 16;
    const h16* gA = A  + (size_t)(m0 + lr) * K + lc;
    const h16* gB = Bw + (size_t)(n0 + lr) * K + lc;
    unsigned aoff[2], boff[2];
#pragma unroll
    for (int p = 0; p < 2; ++p) {
        aoff[p] = (unsigned)(size_t)(&As[p][lr * LDT + lc]);
        boff[p] = (unsigned)(size_t)(&Bs[p][lr * LDT + lc]);
    }

    // prologue: async-fill buffer 0 with K-slice 0
    async_copy_b128(aoff[0],      (unsigned long long)(size_t)(gA));
    async_copy_b128(aoff[0] + 16, (unsigned long long)(size_t)(gA + 8));
    async_copy_b128(boff[0],      (unsigned long long)(size_t)(gB));
    async_copy_b128(boff[0] + 16, (unsigned long long)(size_t)(gB + 8));

    int cur = 0;
    for (int k0 = 0; k0 < K; k0 += BK, cur ^= 1) {
        wait_async0();        // my async fills of buffer `cur` have landed
        __syncthreads();      // everyone's fills done; buffer cur^1 free

        if (k0 + BK < K) {    // async-prefetch next K-slice into cur^1
            const int nxt = cur ^ 1;
            const h16* ga = gA + k0 + BK;
            const h16* gb = gB + k0 + BK;
            async_copy_b128(aoff[nxt],      (unsigned long long)(size_t)(ga));
            async_copy_b128(aoff[nxt] + 16, (unsigned long long)(size_t)(ga + 8));
            async_copy_b128(boff[nxt],      (unsigned long long)(size_t)(gb));
            async_copy_b128(boff[nxt] + 16, (unsigned long long)(size_t)(gb + 8));
        }

        v16h af[2], bf[4];
#pragma unroll
        for (int i = 0; i < 2; ++i)
            af[i] = load_frag_rowmajor(&As[cur][(wm + i * 16 + l16) * LDT], hlf);
#pragma unroll
        for (int j = 0; j < 4; ++j)
            bf[j] = load_frag_rowmajor(&Bs[cur][(wn + j * 16 + l16) * LDT], hlf);
#pragma unroll
        for (int i = 0; i < 2; ++i)
#pragma unroll
            for (int j = 0; j < 4; ++j)
                acc[i][j] = __builtin_amdgcn_wmma_f32_16x16x32_f16(
                    false, af[i], false, bf[j], (short)0, acc[i][j], false, false);
    }

    // D layout: VGPR r -> M = r + 8*hlf ; N = lane%16
#pragma unroll
    for (int i = 0; i < 2; ++i) {
#pragma unroll
        for (int j = 0; j < 4; ++j) {
            int gmb = m0 + wm + i * 16 + hlf * 8;
            int gn  = n0 + wn + j * 16 + l16;
#pragma unroll
            for (int r = 0; r < 8; ++r) {
                float val = acc[i][j][r] * alpha;
                size_t idx = (size_t)(gmb + r) * N + gn;
                if (out_f16) Ch[idx] = (h16)val;
                else         Cf[idx] = val;
            }
        }
    }
}

// -------------------- Flash attention (causal) with WMMA --------------------
// Grid: (S/128, B*H). 8 waves/block; each wave owns 16 query rows.
// qh/kh: (B*S, E) f16 row-major (k pre-scaled by 1/sqrt(Dh)).
// vh: reference quirk -- V element (s, d, h) lives at column d*H + h.
// ctx out: (B*S, E) f32, column h*Dh + d.
#define ATT_LDP 40
__global__ __launch_bounds__(256) void attn_wmma_kernel(
    const h16* __restrict__ qh, const h16* __restrict__ kh,
    const h16* __restrict__ vh, float* __restrict__ ctx) {
    __shared__ h16 Ps[8 * 16 * ATT_LDP];   // per-wave 16x32 P staging

    const int lane = threadIdx.x & 31;
    const int wave = threadIdx.x >> 5;
    const int hlf  = lane >> 4;
    const int l16  = lane & 15;
    const int bh   = blockIdx.y;
    const int b    = bh / H_NUM;
    const int h    = bh % H_NUM;
    const int t0   = blockIdx.x * 128 + wave * 16;
    const size_t rowbase = (size_t)b * S_LEN;

    // resident Q fragments (16 rows x 128 d = 4 K-slices)
    v16h qf[4];
#pragma unroll
    for (int dk = 0; dk < 4; ++dk)
        qf[dk] = load_frag_rowmajor(
            qh + (rowbase + t0 + l16) * E_DIM + h * D_HEAD + dk * 32, hlf);

    v8f  o[8] = {};
    float mrow[8], lrow[8];
#pragma unroll
    for (int r = 0; r < 8; ++r) { mrow[r] = -3.0e38f; lrow[r] = 0.f; }

    h16* myP = &Ps[wave * 16 * ATT_LDP];
    const int s_end = t0 + 16;     // causal: keys s <= t0+15

    for (int s0 = 0; s0 < s_end; s0 += 32) {
        // ---- S = Q * K^T (two 16-key column tiles) ----
        v8f sc[2] = {};
#pragma unroll
        for (int nt = 0; nt < 2; ++nt) {
            int srow = s0 + nt * 16 + l16;
            if (srow > S_LEN - 1) srow = S_LEN - 1;   // masked anyway
            const h16* kb = kh + (rowbase + srow) * E_DIM + h * D_HEAD;
#pragma unroll
            for (int dk = 0; dk < 4; ++dk) {
                v16h kf = load_frag_rowmajor(kb + dk * 32, hlf);
                sc[nt] = __builtin_amdgcn_wmma_f32_16x16x32_f16(
                    false, qf[dk], false, kf, (short)0, sc[nt], false, false);
            }
        }
        // ---- causal mask + online softmax (per-row over 32 cols) ----
#pragma unroll
        for (int r = 0; r < 8; ++r) {
            int t = t0 + r + 8 * hlf;
#pragma unroll
            for (int nt = 0; nt < 2; ++nt) {
                int s = s0 + nt * 16 + l16;
                if (s > t) sc[nt][r] = -3.0e38f;
            }
            float rm = fmaxf(sc[0][r], sc[1][r]);
#pragma unroll
            for (int d = 8; d >= 1; d >>= 1)
                rm = fmaxf(rm, __shfl_xor(rm, d, 32));   // stays in half-group
            float mn = fmaxf(mrow[r], rm);
            float fs = __expf(mrow[r] - mn);
            float p0 = __expf(sc[0][r] - mn);
            float p1 = __expf(sc[1][r] - mn);
            sc[0][r] = p0; sc[1][r] = p1;
            float rs = p0 + p1;
#pragma unroll
            for (int d = 8; d >= 1; d >>= 1)
                rs += __shfl_xor(rs, d, 32);
            lrow[r] = lrow[r] * fs + rs;
            mrow[r] = mn;
#pragma unroll
            for (int j = 0; j < 8; ++j) o[j][r] *= fs;
        }
        // ---- stage P (16x32) through LDS to re-fragment as A operand ----
#pragma unroll
        for (int r = 0; r < 8; ++r) {
            myP[(r + 8 * hlf) * ATT_LDP + l16]      = (h16)sc[0][r];
            myP[(r + 8 * hlf) * ATT_LDP + 16 + l16] = (h16)sc[1][r];
        }
        v16h pf = load_frag_rowmajor(&myP[l16 * ATT_LDP], hlf);
        // ---- O += P * V  (V column for (d,h) is d*H + h, stride H) ----
#pragma unroll
        for (int j = 0; j < 8; ++j) {
            v16h vf;
            int dcol = (j * 16 + l16) * H_NUM + h;
#pragma unroll
            for (int jj = 0; jj < 8; ++jj) {
                int kb = frag_kbase(jj, hlf);
                int s1 = s0 + kb;
                int s2 = s1 + 1;
                if (s1 > S_LEN - 1) s1 = S_LEN - 1;   // p == 0 there
                if (s2 > S_LEN - 1) s2 = S_LEN - 1;
                vf[2 * jj]     = vh[(rowbase + s1) * E_DIM + dcol];
                vf[2 * jj + 1] = vh[(rowbase + s2) * E_DIM + dcol];
            }
            o[j] = __builtin_amdgcn_wmma_f32_16x16x32_f16(
                false, pf, false, vf, (short)0, o[j], false, false);
        }
    }

    // ---- normalize and write ctx (row t, col h*Dh + d) ----
#pragma unroll
    for (int r = 0; r < 8; ++r) {
        float inv = 1.0f / lrow[r];
        int t = t0 + r + 8 * hlf;
#pragma unroll
        for (int j = 0; j < 8; ++j)
            ctx[(rowbase + t) * E_DIM + h * D_HEAD + j * 16 + l16] = o[j][r] * inv;
    }
}

// ------------------------------ epilogues ----------------------------------
__global__ __launch_bounds__(256) void layernorm_f16_kernel(
    const float* __restrict__ X, const float* __restrict__ g,
    const float* __restrict__ bta, h16* __restrict__ out) {
    __shared__ float red[256];
    size_t base = (size_t)blockIdx.x * E_DIM;
    float s = 0.f;
    for (int i = threadIdx.x; i < E_DIM; i += 256) s += X[base + i];
    float mu = block_reduce_sum256(s, red) * (1.0f / E_DIM);
    float v = 0.f;
    for (int i = threadIdx.x; i < E_DIM; i += 256) {
        float d = X[base + i] - mu; v += d * d;
    }
    float var = block_reduce_sum256(v, red) * (1.0f / E_DIM);
    float rstd = rsqrtf(var + 1e-5f);
    for (int i = threadIdx.x; i < E_DIM; i += 256)
        out[base + i] = (h16)((X[base + i] - mu) * rstd * g[i] + bta[i]);
}

// x2 = rmsnorm(x + ao, g); write f32 (for final residual) and f16 (GEMM A)
__global__ __launch_bounds__(256) void add_rmsnorm_kernel(
    const float* __restrict__ X, const float* __restrict__ Ao,
    const float* __restrict__ g, float* __restrict__ outF, h16* __restrict__ outH) {
    __shared__ float red[256];
    size_t base = (size_t)blockIdx.x * E_DIM;
    float s = 0.f;
    for (int i = threadIdx.x; i < E_DIM; i += 256) {
        float v = X[base + i] + Ao[base + i]; s += v * v;
    }
    float ms = block_reduce_sum256(s, red) * (1.0f / E_DIM);
    float r = rsqrtf(ms + 1e-6f);
    for (int i = threadIdx.x; i < E_DIM; i += 256) {
        float v = X[base + i] + Ao[base + i];
        float y = g[i] * (v * r);
        outF[base + i] = y;
        outH[base + i] = (h16)y;
    }
}

__global__ __launch_bounds__(256) void swiglu_kernel(
    const float* __restrict__ w, const float* __restrict__ vv,
    const float* __restrict__ beta, float* __restrict__ g, size_t n) {
    float bb = beta[0];
    size_t i = (size_t)blockIdx.x * 256 + threadIdx.x;
    size_t stride = (size_t)gridDim.x * 256;
    for (; i < n; i += stride) {
        float wv = w[i];
        float sw = wv / (1.0f + __expf(-bb * wv));
        g[i] = sw * vv[i];
    }
}

// out = x2 + rmsnorm(g, gamma)
__global__ __launch_bounds__(256) void rms_residual_out_kernel(
    const float* __restrict__ G, const float* __restrict__ gamma,
    const float* __restrict__ X2, float* __restrict__ out) {
    __shared__ float red[256];
    size_t base = (size_t)blockIdx.x * E_DIM;
    float s = 0.f;
    for (int i = threadIdx.x; i < E_DIM; i += 256) {
        float v = G[base + i]; s += v * v;
    }
    float ms = block_reduce_sum256(s, red) * (1.0f / E_DIM);
    float r = rsqrtf(ms + 1e-6f);
    for (int i = threadIdx.x; i < E_DIM; i += 256)
        out[base + i] = X2[base + i] + gamma[i] * (G[base + i] * r);
}

// ------------------------------- launcher ----------------------------------
extern "C" void kernel_launch(void* const* d_in, const int* in_sizes, int n_in,
                              void* d_out, int out_size, void* d_ws, size_t ws_size,
                              hipStream_t stream) {
    (void)in_sizes; (void)n_in; (void)out_size; (void)ws_size;
    const float* x      = (const float*)d_in[0];
    const float* Wq     = (const float*)d_in[1];
    const float* Wk     = (const float*)d_in[2];
    const float* Wv     = (const float*)d_in[3];
    const float* Wo     = (const float*)d_in[4];
    const float* ln_g   = (const float*)d_in[5];
    const float* ln_b   = (const float*)d_in[6];
    const float* rms_g  = (const float*)d_in[7];
    const float* W0     = (const float*)d_in[8];
    const float* swW    = (const float*)d_in[9];
    const float* swV    = (const float*)d_in[10];
    const float* sw_b   = (const float*)d_in[11];
    const float* mrms_g = (const float*)d_in[12];

    const size_t S16b = (size_t)M_TOT * E_DIM * sizeof(h16);
    const size_t S32b = (size_t)M_TOT * E_DIM * sizeof(float);
    const size_t Wcnt = (size_t)E_DIM * E_DIM;

    char* ws = (char*)d_ws;
    size_t off = 0;
    h16*   xh    = (h16*)(ws + off);  off += S16b;
    h16*   qhp   = (h16*)(ws + off);  off += S16b;
    h16*   khp   = (h16*)(ws + off);  off += S16b;
    h16*   vhp   = (h16*)(ws + off);  off += S16b;
    float* ctx   = (float*)(ws + off); off += S32b;
    h16*   cth   = (h16*)(ws + off);  off += S16b;
    float* ao    = (float*)(ws + off); off += S32b;
    float* x2f   = (float*)(ws + off); off += S32b;
    h16*   x2h   = (h16*)(ws + off);  off += S16b;
    h16*   hh    = (h16*)(ws + off);  off += S16b;
    float* wbuf  = (float*)(ws + off); off += S32b;
    float* vvbuf = (float*)(ws + off); off += S32b;
    float* gbuf  = (float*)(ws + off); off += S32b;
    h16*   wtab  = (h16*)(ws + off);
    h16* Wqh = wtab + 0 * Wcnt;
    h16* Wkh = wtab + 1 * Wcnt;
    h16* Wvh = wtab + 2 * Wcnt;
    h16* Woh = wtab + 3 * Wcnt;
    h16* W0h = wtab + 4 * Wcnt;
    h16* sWh = wtab + 5 * Wcnt;
    h16* sVh = wtab + 6 * Wcnt;

    const int nAct = M_TOT * E_DIM;   // 8M
    const int nW   = E_DIM * E_DIM;   // 4M
    cvt_f32_to_f16_kernel<<<2048, 256, 0, stream>>>(x,   xh,  nAct);
    cvt_f32_to_f16_kernel<<<1024, 256, 0, stream>>>(Wq,  Wqh, nW);
    cvt_f32_to_f16_kernel<<<1024, 256, 0, stream>>>(Wk,  Wkh, nW);
    cvt_f32_to_f16_kernel<<<1024, 256, 0, stream>>>(Wv,  Wvh, nW);
    cvt_f32_to_f16_kernel<<<1024, 256, 0, stream>>>(Wo,  Woh, nW);
    cvt_f32_to_f16_kernel<<<1024, 256, 0, stream>>>(W0,  W0h, nW);
    cvt_f32_to_f16_kernel<<<1024, 256, 0, stream>>>(swW, sWh, nW);
    cvt_f32_to_f16_kernel<<<1024, 256, 0, stream>>>(swV, sVh, nW);

    const dim3 gg(E_DIM / BN, M_TOT / BM);      // (16, 32)
    const float kscale = 0.08838834764831845f;  // 1/sqrt(128)

    // q, k (pre-scaled), v in f16 for attention
    gemm_wmma_kernel<<<gg, 256, 0, stream>>>(xh, Wqh, nullptr, qhp, M_TOT, E_DIM, E_DIM, 1.0f,   1);
    gemm_wmma_kernel<<<gg, 256, 0, stream>>>(xh, Wkh, nullptr, khp, M_TOT, E_DIM, E_DIM, kscale, 1);
    gemm_wmma_kernel<<<gg, 256, 0, stream>>>(xh, Wvh, nullptr, vhp, M_TOT, E_DIM, E_DIM, 1.0f,   1);

    attn_wmma_kernel<<<dim3(S_LEN / 128, B_SZ * H_NUM), 256, 0, stream>>>(qhp, khp, vhp, ctx);

    layernorm_f16_kernel<<<M_TOT, 256, 0, stream>>>(ctx, ln_g, ln_b, cth);
    gemm_wmma_kernel<<<gg, 256, 0, stream>>>(cth, Woh, ao, nullptr, M_TOT, E_DIM, E_DIM, 1.0f, 0);

    add_rmsnorm_kernel<<<M_TOT, 256, 0, stream>>>(x, ao, rms_g, x2f, x2h);

    gemm_wmma_kernel<<<gg, 256, 0, stream>>>(x2h, W0h, nullptr, hh, M_TOT, E_DIM, E_DIM, 1.0f, 1);
    gemm_wmma_kernel<<<gg, 256, 0, stream>>>(hh, sWh, wbuf,  nullptr, M_TOT, E_DIM, E_DIM, 1.0f, 0);
    gemm_wmma_kernel<<<gg, 256, 0, stream>>>(hh, sVh, vvbuf, nullptr, M_TOT, E_DIM, E_DIM, 1.0f, 0);

    swiglu_kernel<<<2048, 256, 0, stream>>>(wbuf, vvbuf, sw_b, gbuf, (size_t)M_TOT * E_DIM);
    rms_residual_out_kernel<<<M_TOT, 256, 0, stream>>>(gbuf, mrms_g, x2f, (float*)d_out);
}
